// Transformer_14894946583012
// MI455X (gfx1250) — matrix operand order repaired and checked by reading, not compiled, not run on hardware
//
#include <hip/hip_runtime.h>

// ---------------------------------------------------------------------------
// Types for CDNA5 (gfx1250) WMMA: wave32, v_wmma_f32_16x16x32_bf16
// ---------------------------------------------------------------------------
typedef __bf16 bf16;
typedef bf16  v16bf __attribute__((ext_vector_type(16)));
typedef float v8f   __attribute__((ext_vector_type(8)));
typedef unsigned short u16;
typedef u16 u16x8 __attribute__((ext_vector_type(8)));

#define B_   2
#define S_   1024
#define U_   1024
#define H_   16
#define D_   64
#define NT_  32000
#define MEM_ 576
#define FF_  4096
#define KL_  1600   // MEM_ + S_

__device__ __forceinline__ u16 f2bf(float f) {
  unsigned u = __float_as_uint(f);
  u += 0x7fffu + ((u >> 16) & 1u);           // round-to-nearest-even
  return (u16)(u >> 16);
}

union Frag {
  v16bf v;
  u16x8 u[2];
  u16   s[16];
};

// A-fragment (16x32 bf16, row-major source): caller passes pointer at
// row*ld + k0 + 8*half.  Elements e<8 -> K=k0+8h+e ; e>=8 -> K=k0+8h+16+(e-8)
__device__ __forceinline__ v16bf load_a_frag(const u16* p) {
  Frag f;
  f.u[0] = *(const u16x8*)(p);
  f.u[1] = *(const u16x8*)(p + 16);
  return f.v;
}
// B-fragment for sources stored as B^T row-major (lane holds one column of B,
// i.e. one row of B^T); e<8 -> K=k0+8h+e ; e>=8 -> K=k0+8h+16+(e-8).
// Caller passes pointer at col_row*ld + k0 + 8*half.
__device__ __forceinline__ v16bf load_bt_frag(const u16* p) {
  Frag f;
  f.u[0] = *(const u16x8*)(p);
  f.u[1] = *(const u16x8*)(p + 16);
  return f.v;
}

__device__ __forceinline__ v8f wmma_bf16(v16bf a, v16bf b, v8f c) {
  return __builtin_amdgcn_wmma_f32_16x16x32_bf16(false, a, false, b, (short)0,
                                                 c, false, false);
}

// ---------------------------------------------------------------------------
// CDNA5 async global->LDS staging (ASYNCcnt path), with fallback to VGPR
// round-trip staging when the builtins are not declared by the toolchain.
// Builtin signature (from clang diagnostic): (global v4i*, local v4i*, Ii, Ii).
// Pointers are built via integer casts: low 32 bits of a generic LDS address
// are the LDS byte offset.
// ---------------------------------------------------------------------------
#if __has_builtin(__builtin_amdgcn_global_load_async_to_lds_b128)
#define HAVE_ASYNC_LDS 1
typedef int v4i __attribute__((ext_vector_type(4)));
typedef __attribute__((address_space(1))) v4i* gv4p;
typedef __attribute__((address_space(3))) v4i* lv4p;
__device__ __forceinline__ void async_copy16(const u16* g, u16* l) {
  __builtin_amdgcn_global_load_async_to_lds_b128(
      (gv4p)(unsigned long long)g,
      (lv4p)(unsigned)(unsigned long long)l, 0, 0);
}
#else
#define HAVE_ASYNC_LDS 0
#endif

__device__ __forceinline__ void wait_async0() {
#if __has_builtin(__builtin_amdgcn_s_wait_asynccnt)
  __builtin_amdgcn_s_wait_asynccnt(0);
#else
  asm volatile("s_wait_asynccnt 0" ::: "memory");
#endif
}

// ---------------------------------------------------------------------------
// Tiled transpose + fp32->bf16:  src (K x N fp32)  ->  dst (N x K bf16)
// ---------------------------------------------------------------------------
__global__ __launch_bounds__(256) void k_cvt_t(const float* __restrict__ src,
                                               u16* __restrict__ dst,
                                               int K, int N) {
  __shared__ u16 tile[32][33];
  int n0 = blockIdx.x * 32, k0 = blockIdx.y * 32;
  int tx = threadIdx.x & 31, ty = threadIdx.x >> 5;   // ty in 0..7
#pragma unroll
  for (int i = 0; i < 4; i++) {
    int k = k0 + ty + i * 8;
    tile[ty + i * 8][tx] = f2bf(src[(size_t)k * N + n0 + tx]);
  }
  __syncthreads();
#pragma unroll
  for (int i = 0; i < 4; i++) {
    int n = n0 + ty + i * 8;
    dst[(size_t)n * K + k0 + tx] = tile[tx][ty + i * 8];
  }
}

// ---------------------------------------------------------------------------
// Embedding gather (* sqrt(U)=32) -> x fp32, x bf16, and tail of `full` bf16
// ---------------------------------------------------------------------------
__global__ __launch_bounds__(256) void k_embed(const int* __restrict__ tokens,
                                               const float* __restrict__ tab,
                                               float* __restrict__ x,
                                               u16* __restrict__ xh,
                                               u16* __restrict__ fullh) {
  int bs = blockIdx.x;
  int b = bs / S_, s = bs % S_;
  int tok = tokens[bs];
  const float* row = tab + (size_t)tok * U_;
  size_t xo = (size_t)bs * U_;
  size_t fo = ((size_t)(b * KL_ + MEM_ + s)) * U_;
  for (int u = threadIdx.x; u < U_; u += blockDim.x) {
    float v = row[u] * 32.0f;
    x[xo + u] = v;
    u16 hv = f2bf(v);
    xh[xo + u] = hv;
    fullh[fo + u] = hv;
  }
}

__global__ __launch_bounds__(256) void k_memcvt(const float* __restrict__ mem,
                                                u16* __restrict__ fullh) {
  int bm = blockIdx.x;
  int b = bm / MEM_, m = bm % MEM_;
  size_t src = (size_t)bm * U_;
  size_t dst = ((size_t)(b * KL_ + m)) * U_;
  for (int u = threadIdx.x; u < U_; u += blockDim.x)
    fullh[dst + u] = f2bf(mem[src + u]);
}

// ---------------------------------------------------------------------------
// Transformer-XL sinusoidal relative positions (descending), bf16
// ---------------------------------------------------------------------------
__global__ __launch_bounds__(256) void k_relpos(u16* __restrict__ relh) {
  int p = blockIdx.x;
  float pos = (float)(KL_ - 1 - p);
  for (int u = threadIdx.x; u < U_; u += blockDim.x) {
    int j = (u < U_ / 2) ? u : (u - U_ / 2);
    float freq = __expf(-(float)j * (9.2103403719761836f / 512.0f)); // 10000^-(j/512)
    float ang = pos * freq;
    float v = (u < U_ / 2) ? sinf(ang) : cosf(ang);
    relh[(size_t)p * U_ + u] = f2bf(v);
  }
}

// ---------------------------------------------------------------------------
// q + bias_context / q + bias_relative -> bf16
// ---------------------------------------------------------------------------
__global__ __launch_bounds__(256) void k_qsplit(const float* __restrict__ q,
                                                const float* __restrict__ bc,
                                                const float* __restrict__ br,
                                                u16* __restrict__ qch,
                                                u16* __restrict__ qrh) {
  size_t i = (size_t)blockIdx.x * blockDim.x + threadIdx.x;
  int u = (int)(i & (U_ - 1));
  float v = q[i];
  qch[i] = f2bf(v + bc[u]);
  qrh[i] = f2bf(v + br[u]);
}

// kv (B*KL, 2U) fp32 -> k_h (B,KL,U) bf16  and  v^T (B,H,D,KL) bf16
__global__ __launch_bounds__(256) void k_kvsplit(const float* __restrict__ kv,
                                                 u16* __restrict__ kh,
                                                 u16* __restrict__ vt) {
  size_t i = (size_t)blockIdx.x * blockDim.x + threadIdx.x;
  size_t rowi = i / U_;
  int u = (int)(i & (U_ - 1));
  int b = (int)(rowi / KL_);
  int j = (int)(rowi % KL_);
  int h = u >> 6, d = u & 63;
  size_t src = rowi * (2 * U_);
  kh[i] = f2bf(kv[src + u]);
  vt[(((size_t)b * H_ + h) * D_ + d) * KL_ + j] = f2bf(kv[src + U_ + u]);
}

// ---------------------------------------------------------------------------
// bf16 WMMA GEMM:  C[M,N] = A[M,K] @ B[K,N] + bias, B given TRANSPOSED (N x K).
// Block: 256 threads = 8 waves (2x4); tile 64(M) x 256(N); K-step 64.
// Wave tile 32x64 = 2x4 WMMA accumulators; 16 WMMA per wave per K-step.
// Staging uses CDNA5 async global->LDS copies when available.
// ---------------------------------------------------------------------------
#define GM 64
#define GN 256
template <int RELU, int WC, int WH>
__global__ __launch_bounds__(256) void k_gemm(const u16* __restrict__ A,
                                              const u16* __restrict__ Bt,
                                              const float* __restrict__ bias,
                                              float* __restrict__ C,
                                              u16* __restrict__ Ch,
                                              int M, int N, int K) {
  __shared__ u16 As[GM * 64];   //  8 KB
  __shared__ u16 Bs[GN * 64];   // 32 KB
  int t = threadIdx.x;
  int lane = t & 31, w = t >> 5;
  int half = lane >> 4, col = lane & 15;
  int wm = w >> 2, wn = w & 3;
  int m0 = blockIdx.y * GM, n0 = blockIdx.x * GN;
  (void)M;

  v8f acc[2][4];
#pragma unroll
  for (int a = 0; a < 2; a++)
#pragma unroll
    for (int b = 0; b < 4; b++) {
      v8f z = {0.f, 0.f, 0.f, 0.f, 0.f, 0.f, 0.f, 0.f};
      acc[a][b] = z;
    }

  for (int k0 = 0; k0 < K; k0 += 64) {
    // stage A: 64x64 = 4096 elems, 16 per thread (two 16B chunks of one row)
    // stage B^T: 256 rows x 64, one full row (128B contiguous) per thread
    const u16* ga = &A[(size_t)(m0 + (t >> 2)) * K + k0 + (t & 3) * 16];
    u16* da = &As[(t >> 2) * 64 + (t & 3) * 16];
    const u16* gb = &Bt[(size_t)(n0 + t) * K + k0];
    u16* db = &Bs[t * 64];
#if HAVE_ASYNC_LDS
    async_copy16(ga, da);
    async_copy16(ga + 8, da + 8);
#pragma unroll
    for (int q = 0; q < 8; q++) async_copy16(gb + q * 8, db + q * 8);
    wait_async0();
#else
    *(u16x8*)(da) = *(const u16x8*)(ga);
    *(u16x8*)(da + 8) = *(const u16x8*)(ga + 8);
#pragma unroll
    for (int q = 0; q < 8; q++)
      *(u16x8*)(db + q * 8) = *(const u16x8*)(gb + q * 8);
#endif
    __syncthreads();
#pragma unroll
    for (int kk = 0; kk < 2; kk++) {
      v16bf af[2], bfr[4];
#pragma unroll
      for (int mt = 0; mt < 2; mt++)
        af[mt] =
            load_a_frag(&As[(wm * 32 + mt * 16 + col) * 64 + kk * 32 + half * 8]);
#pragma unroll
      for (int nt = 0; nt < 4; nt++)
        bfr[nt] = load_bt_frag(
            &Bs[(wn * 64 + nt * 16 + col) * 64 + kk * 32 + 8 * half]);
#pragma unroll
      for (int mt = 0; mt < 2; mt++)
#pragma unroll
        for (int nt = 0; nt < 4; nt++)
          acc[mt][nt] = wmma_bf16(af[mt], bfr[nt], acc[mt][nt]);
    }
    __syncthreads();
  }

#pragma unroll
  for (int mt = 0; mt < 2; mt++)
#pragma unroll
    for (int nt = 0; nt < 4; nt++)
#pragma unroll
      for (int v = 0; v < 8; v++) {
        int row = m0 + wm * 32 + mt * 16 + v + 8 * half;
        int cn = n0 + wn * 64 + nt * 16 + col;
        float val = acc[mt][nt][v] + bias[cn];
        if (RELU) val = fmaxf(val, 0.f);
        if (WC) C[(size_t)row * N + cn] = val;
        if (WH) Ch[(size_t)row * N + cn] = f2bf(val);
      }
}

// ---------------------------------------------------------------------------
// One 32-key block of the fused rel-pos flash attention.
// MASKED=false: whole block is causally valid (j0+31 <= i0+MEM) -> no compares,
// no r-row clamp.  MASKED=true: boundary blocks with per-element masking.
// ---------------------------------------------------------------------------
template <bool MASKED>
__device__ __forceinline__ void attn_kblock(
    int j0, int i0, int half, int col,
    const u16* __restrict__ kbase,   // kh + (b*KL)*U + h*D
    const u16* __restrict__ rbase,   // rh + h*D
    const u16* __restrict__ vbase,   // vt + (b*H+h)*D*KL
    v16bf qc0, v16bf qc1, v16bf qr0, v16bf qr1,
    v8f O[4], float* mrow, float* srow,
    float* __restrict__ GbW, u16* __restrict__ PsW) {
  // ---- content scores: two 16x16 j-tiles
  v8f Sc[2];
#pragma unroll
  for (int tt = 0; tt < 2; tt++) {
    v8f z = {0.f, 0.f, 0.f, 0.f, 0.f, 0.f, 0.f, 0.f};
    Sc[tt] = z;
    const u16* kp = kbase + (size_t)(j0 + 16 * tt + col) * U_ + 8 * half;
    Sc[tt] = wmma_bf16(qc0, load_bt_frag(kp), Sc[tt]);
    Sc[tt] = wmma_bf16(qc1, load_bt_frag(kp + 32), Sc[tt]);
  }

  // ---- positional band G[i, m], m in [m_min, m_min+48)
  int m_min = j0 - i0 + 1008;
#pragma unroll
  for (int mt = 0; mt < 3; mt++) {
    int m = m_min + 16 * mt + col;
    if (MASKED) {
      if (m > KL_ - 1) m = KL_ - 1;  // masked region only
    }
    const u16* rp = rbase + (size_t)m * U_ + 8 * half;
    v8f G = {0.f, 0.f, 0.f, 0.f, 0.f, 0.f, 0.f, 0.f};
    G = wmma_bf16(qr0, load_bt_frag(rp), G);
    G = wmma_bf16(qr1, load_bt_frag(rp + 32), G);
#pragma unroll
    for (int v = 0; v < 8; v++)
      GbW[(v + 8 * half) * 48 + 16 * mt + col] = G[v];
  }
  asm volatile("s_wait_dscnt 0" ::: "memory");

  // ---- combine, scale, mask; row-max over the 32 keys
  float Sf0[8], Sf1[8], pm[8];
#pragma unroll
  for (int v = 0; v < 8; v++) {
    int iloc = v + 8 * half;
    float ar0 = GbW[iloc * 48 + (col - iloc + 15)];
    float ar1 = GbW[iloc * 48 + (16 + col - iloc + 15)];
    float s0 = (Sc[0][v] + ar0) * 0.125f;
    float s1 = (Sc[1][v] + ar1) * 0.125f;
    if (MASKED) {
      int iabs = i0 + iloc;
      if (j0 + col > iabs + MEM_) s0 = -3e38f;
      if (j0 + 16 + col > iabs + MEM_) s1 = -3e38f;
    }
    Sf0[v] = s0; Sf1[v] = s1;
    pm[v] = fmaxf(s0, s1);
  }
#pragma unroll
  for (int v = 0; v < 8; v++) {
#pragma unroll
    for (int off = 8; off; off >>= 1)
      pm[v] = fmaxf(pm[v], __shfl_xor(pm[v], off, 32));
  }

  // ---- online softmax update
#pragma unroll
  for (int v = 0; v < 8; v++) {
    float mnew = fmaxf(mrow[v], pm[v]);
    float alpha = __expf(mrow[v] - mnew);
    float p0 = __expf(Sf0[v] - mnew);
    float p1 = __expf(Sf1[v] - mnew);
    Sf0[v] = p0; Sf1[v] = p1;
    float ps = p0 + p1;
#pragma unroll
    for (int off = 8; off; off >>= 1) ps += __shfl_xor(ps, off, 32);
    srow[v] = srow[v] * alpha + ps;
    mrow[v] = mnew;
    O[0][v] *= alpha; O[1][v] *= alpha; O[2][v] *= alpha; O[3][v] *= alpha;
  }

  // ---- P (16x32) to LDS as bf16, reload as A-fragment
#pragma unroll
  for (int v = 0; v < 8; v++) {
    PsW[(v + 8 * half) * 32 + col] = f2bf(Sf0[v]);
    PsW[(v + 8 * half) * 32 + 16 + col] = f2bf(Sf1[v]);
  }
  asm volatile("s_wait_dscnt 0" ::: "memory");
  v16bf pa = load_a_frag(&PsW[col * 32 + half * 8]);

  // ---- P @ V with V^T: contiguous B-fragments
#pragma unroll
  for (int dt = 0; dt < 4; dt++) {
    const u16* vp = vbase + (size_t)(dt * 16 + col) * KL_ + j0 + 8 * half;
    O[dt] = wmma_bf16(pa, load_bt_frag(vp), O[dt]);
  }
}

// ---------------------------------------------------------------------------
// Fused rel-pos flash attention.  Wave = one (b, h, 16-row query tile).
// Unmasked main loop + masked tail (only boundary blocks pay mask cost).
// ---------------------------------------------------------------------------
__global__ __launch_bounds__(128) void k_attn(const u16* __restrict__ qc,
                                              const u16* __restrict__ qr,
                                              const u16* __restrict__ kh,
                                              const u16* __restrict__ vt,
                                              const u16* __restrict__ rh,
                                              u16* __restrict__ outh) {
  __shared__ float Gb[4][16 * 48];
  __shared__ u16 Ps[4][16 * 32];
  int lane = threadIdx.x & 31, w = threadIdx.x >> 5;
  int half = lane >> 4, col = lane & 15;
  int wg = blockIdx.x * 4 + w;
  int b = wg / (H_ * 64);
  int rem = wg % (H_ * 64);
  int h = rem / 64;
  int i0 = (rem % 64) * 16;

  size_t qoff = ((size_t)(b * S_ + i0 + col)) * U_ + h * D_;
  v16bf qc0 = load_a_frag(qc + qoff + half * 8);
  v16bf qc1 = load_a_frag(qc + qoff + 32 + half * 8);
  v16bf qr0 = load_a_frag(qr + qoff + half * 8);
  v16bf qr1 = load_a_frag(qr + qoff + 32 + half * 8);

  const u16* kbase = kh + (size_t)b * KL_ * U_ + h * D_;
  const u16* rbase = rh + h * D_;
  const u16* vbase = vt + ((size_t)(b * H_ + h)) * D_ * KL_;
  float* GbW = &Gb[w][0];
  u16* PsW = &Ps[w][0];

  v8f O[4];
#pragma unroll
  for (int dt = 0; dt < 4; dt++) {
    v8f z = {0.f, 0.f, 0.f, 0.f, 0.f, 0.f, 0.f, 0.f};
    O[dt] = z;
  }
  float mrow[8], srow[8];
#pragma unroll
  for (int v = 0; v < 8; v++) { mrow[v] = -3e38f; srow[v] = 0.f; }

  int kb_full = (i0 + MEM_ - 31) / 32 + 1;   // fully-valid key blocks
  int nblocks = (i0 + 15 + MEM_) / 32 + 1;   // all contributing key blocks
  for (int kb = 0; kb < kb_full; ++kb)
    attn_kblock<false>(kb * 32, i0, half, col, kbase, rbase, vbase, qc0, qc1,
                       qr0, qr1, O, mrow, srow, GbW, PsW);
  for (int kb = kb_full; kb < nblocks; ++kb)
    attn_kblock<true>(kb * 32, i0, half, col, kbase, rbase, vbase, qc0, qc1,
                      qr0, qr1, O, mrow, srow, GbW, PsW);

  // ---- epilogue: normalize + store bf16 (B,S,U)
#pragma unroll
  for (int v = 0; v < 8; v++) {
    float inv = 1.0f / srow[v];
    size_t base = ((size_t)(b * S_ + i0 + v + 8 * half)) * U_ + h * D_ + col;
#pragma unroll
    for (int dt = 0; dt < 4; dt++)
      outh[base + dt * 16] = f2bf(O[dt][v] * inv);
  }
}

// ---------------------------------------------------------------------------
// h = LayerNorm(a + b) * g + beta ; optional fp32 and bf16 outputs
// ---------------------------------------------------------------------------
__global__ __launch_bounds__(256) void k_addln(const float* __restrict__ a,
                                               const float* __restrict__ bsrc,
                                               const float* __restrict__ g,
                                               const float* __restrict__ bb,
                                               float* __restrict__ out32,
                                               u16* __restrict__ outh) {
  int row = blockIdx.x;
  size_t base = (size_t)row * U_;
  __shared__ float r1[256], r2[256];
  float vals[4];
  float s = 0.f, s2 = 0.f;
#pragma unroll
  for (int c = 0; c < 4; c++) {
    int u = threadIdx.x + 256 * c;
    float v = a[base + u] + bsrc[base + u];
    vals[c] = v; s += v; s2 += v * v;
  }
  r1[threadIdx.x] = s; r2[threadIdx.x] = s2;
  __syncthreads();
  for (int st = 128; st > 0; st >>= 1) {
    if (threadIdx.x < st) {
      r1[threadIdx.x] += r1[threadIdx.x + st];
      r2[threadIdx.x] += r2[threadIdx.x + st];
    }
    __syncthreads();
  }
  float mean = r1[0] * (1.0f / U_);
  float var = r2[0] * (1.0f / U_) - mean * mean;
  float rstd = rsqrtf(var + 1e-5f);
#pragma unroll
  for (int c = 0; c < 4; c++) {
    int u = threadIdx.x + 256 * c;
    float o = (vals[c] - mean) * rstd * g[u] + bb[u];
    if (out32) out32[base + u] = o;
    if (outh) outh[base + u] = f2bf(o);
  }
}

// ---------------------------------------------------------------------------
// Row softmax in place (N = 32000)
// ---------------------------------------------------------------------------
__global__ __launch_bounds__(256) void k_softmax(float* __restrict__ x, int N) {
  size_t base = (size_t)blockIdx.x * N;
  __shared__ float red[256];
  float m = -3e38f;
  for (int i = threadIdx.x; i < N; i += 256) m = fmaxf(m, x[base + i]);
  red[threadIdx.x] = m;
  __syncthreads();
  for (int st = 128; st > 0; st >>= 1) {
    if (threadIdx.x < st)
      red[threadIdx.x] = fmaxf(red[threadIdx.x], red[threadIdx.x + st]);
    __syncthreads();
  }
  m = red[0];
  __syncthreads();
  float sum = 0.f;
  for (int i = threadIdx.x; i < N; i += 256) {
    float e = __expf(x[base + i] - m);
    x[base + i] = e;
    sum += e;
  }
  red[threadIdx.x] = sum;
  __syncthreads();
  for (int st = 128; st > 0; st >>= 1) {
    if (threadIdx.x < st) red[threadIdx.x] += red[threadIdx.x + st];
    __syncthreads();
  }
  float inv = 1.0f / red[0];
  for (int i = threadIdx.x; i < N; i += 256) x[base + i] *= inv;
}

// ---------------------------------------------------------------------------
// Host orchestration
// ---------------------------------------------------------------------------
extern "C" void kernel_launch(void* const* d_in, const int* in_sizes, int n_in,
                              void* d_out, int out_size, void* d_ws,
                              size_t ws_size, hipStream_t stream) {
  (void)in_sizes; (void)n_in; (void)out_size; (void)ws_size;

  const int* tokens = (const int*)d_in[0];
  const float* memory = (const float*)d_in[1];
  const float* embed = (const float*)d_in[2];
  const float* Wq = (const float*)d_in[3];
  const float* Wkv = (const float*)d_in[4];
  const float* Wr = (const float*)d_in[5];
  const float* Wo = (const float*)d_in[6];
  const float* bq = (const float*)d_in[7];
  const float* bkv = (const float*)d_in[8];
  const float* br = (const float*)d_in[9];
  const float* bo = (const float*)d_in[10];
  const float* bctx = (const float*)d_in[11];
  const float* brel = (const float*)d_in[12];
  const float* ln1g = (const float*)d_in[13];
  const float* ln1b = (const float*)d_in[14];
  const float* W1 = (const float*)d_in[15];
  const float* b1 = (const float*)d_in[16];
  const float* W2 = (const float*)d_in[17];
  const float* b2 = (const float*)d_in[18];
  const float* ln2g = (const float*)d_in[19];
  const float* ln2b = (const float*)d_in[20];
  const float* Wout = (const float*)d_in[21];
  const float* bout = (const float*)d_in[22];
  float* out = (float*)d_out;

  char* ws = (char*)d_ws;
  size_t cur = 0;
  auto alloc = [&](size_t elems, size_t esz) -> void* {
    void* p = ws + cur;
    cur += ((elems * esz + 255) & ~(size_t)255);
    return p;
  };

  // transposed bf16 weights (N x K)
  u16* Wq_t = (u16*)alloc((size_t)U_ * U_, 2);
  u16* Wkv_t = (u16*)alloc((size_t)U_ * 2 * U_, 2);
  u16* Wr_t = (u16*)alloc((size_t)U_ * U_, 2);
  u16* Wo_t = (u16*)alloc((size_t)U_ * U_, 2);
  u16* W1_t = (u16*)alloc((size_t)U_ * FF_, 2);
  u16* W2_t = (u16*)alloc((size_t)FF_ * U_, 2);
  u16* Wout_t = (u16*)alloc((size_t)U_ * NT_, 2);

  float* xf = (float*)alloc((size_t)B_ * S_ * U_, 4);
  u16* xh = (u16*)alloc((size_t)B_ * S_ * U_, 2);
  u16* fullh = (u16*)alloc((size_t)B_ * KL_ * U_, 2);
  u16* relh = (u16*)alloc((size_t)KL_ * U_, 2);

  float* qf = (float*)alloc((size_t)B_ * S_ * U_, 4);
  u16* qch = (u16*)alloc((size_t)B_ * S_ * U_, 2);
  u16* qrh = (u16*)alloc((size_t)B_ * S_ * U_, 2);
  float* kvf = (float*)alloc((size_t)B_ * KL_ * 2 * U_, 4);
  u16* khh = (u16*)alloc((size_t)B_ * KL_ * U_, 2);
  u16* vth = (u16*)alloc((size_t)B_ * KL_ * U_, 2);   // V^T (b,h,d,KL)
  u16* rhh = (u16*)alloc((size_t)KL_ * U_, 2);

  u16* attnh = (u16*)alloc((size_t)B_ * S_ * U_, 2);
  float* of = (float*)alloc((size_t)B_ * S_ * U_, 4);
  float* h1f = (float*)alloc((size_t)B_ * S_ * U_, 4);
  u16* h1h = (u16*)alloc((size_t)B_ * S_ * U_, 2);
  u16* ffh = (u16*)alloc((size_t)B_ * S_ * FF_, 2);
  float* f2f = (float*)alloc((size_t)B_ * S_ * U_, 4);
  u16* h2h = (u16*)alloc((size_t)B_ * S_ * U_, 2);

  auto cvt_t = [&](const float* s, u16* d, int K, int N) {
    k_cvt_t<<<dim3(N / 32, K / 32), dim3(256), 0, stream>>>(s, d, K, N);
  };
  cvt_t(Wq, Wq_t, U_, U_);
  cvt_t(Wkv, Wkv_t, U_, 2 * U_);
  cvt_t(Wr, Wr_t, U_, U_);
  cvt_t(Wo, Wo_t, U_, U_);
  cvt_t(W1, W1_t, U_, FF_);
  cvt_t(W2, W2_t, FF_, U_);
  cvt_t(Wout, Wout_t, U_, NT_);

  k_embed<<<dim3(B_ * S_), dim3(256), 0, stream>>>(tokens, embed, xf, xh, fullh);
  k_memcvt<<<dim3(B_ * MEM_), dim3(256), 0, stream>>>(memory, fullh);
  k_relpos<<<dim3(KL_), dim3(256), 0, stream>>>(relh);

  auto gemm = [&](const u16* A, const u16* Bt, const float* bias, float* C,
                  u16* Ch, int M, int N, int K, int relu) {
    dim3 g(N / GN, M / GM), blk(256);
    if (relu)
      k_gemm<1, 0, 1><<<g, blk, 0, stream>>>(A, Bt, bias, C, Ch, M, N, K);
    else if (Ch)
      k_gemm<0, 0, 1><<<g, blk, 0, stream>>>(A, Bt, bias, C, Ch, M, N, K);
    else
      k_gemm<0, 1, 0><<<g, blk, 0, stream>>>(A, Bt, bias, C, Ch, M, N, K);
  };

  gemm(xh, Wq_t, bq, qf, nullptr, B_ * S_, U_, U_, 0);              // q
  gemm(fullh, Wkv_t, bkv, kvf, nullptr, B_ * KL_, 2 * U_, U_, 0);   // kv
  gemm(relh, Wr_t, br, nullptr, rhh, KL_, U_, U_, 0);               // r

  k_qsplit<<<dim3((B_ * S_ * U_) / 256), dim3(256), 0, stream>>>(qf, bctx, brel,
                                                                 qch, qrh);
  k_kvsplit<<<dim3((B_ * KL_ * U_) / 256), dim3(256), 0, stream>>>(kvf, khh, vth);

  k_attn<<<dim3((B_ * H_ * (S_ / 16)) / 4), dim3(128), 0, stream>>>(
      qch, qrh, khh, vth, rhh, attnh);

  gemm(attnh, Wo_t, bo, of, nullptr, B_ * S_, U_, U_, 0);           // out @ Wo
  k_addln<<<dim3(B_ * S_), dim3(256), 0, stream>>>(xf, of, ln1g, ln1b, h1f, h1h);
  gemm(h1h, W1_t, b1, nullptr, ffh, B_ * S_, FF_, U_, 1);           // FF1+relu
  gemm(ffh, W2_t, b2, f2f, nullptr, B_ * S_, U_, FF_, 0);           // FF2
  k_addln<<<dim3(B_ * S_), dim3(256), 0, stream>>>(h1f, f2f, ln2g, ln2b, nullptr,
                                                   h2h);
  gemm(h2h, Wout_t, bout, out, nullptr, B_ * S_, NT_, U_, 0);       // logits
  k_softmax<<<dim3(B_ * S_), dim3(256), 0, stream>>>(out, NT_);
}
